// GeometricAttentionLayer_24876450579070
// MI455X (gfx1250) — matrix operand order repaired and checked by reading, not compiled, not run on hardware
//
#include <hip/hip_runtime.h>
#include <hip/hip_bf16.h>
#include <math.h>

// ---------------------------------------------------------------------------
// GeometricAttentionLayer for MI455X (gfx1250): f16 WMMA everywhere,
// TDM (tensor DMA) double-buffered GEMM (128x128 block tile, 8 WMMA/wave/step),
// flash-attention, f32 accumulation + LN.
// ---------------------------------------------------------------------------

typedef __attribute__((ext_vector_type(16))) _Float16 v16h;
typedef __attribute__((ext_vector_type(8)))  float    v8f;
typedef __attribute__((ext_vector_type(4)))  float    f32x4;
typedef __attribute__((ext_vector_type(4)))  unsigned int u32x4;
typedef __attribute__((ext_vector_type(4)))  int      i32x4;
typedef __attribute__((ext_vector_type(8)))  int      i32x8;

#if defined(__AMDGCN__) && __has_builtin(__builtin_amdgcn_tensor_load_to_lds) && \
    __has_builtin(__builtin_amdgcn_s_wait_tensorcnt)
#define USE_TDM 1
#else
#define USE_TDM 0
#endif

// Load a 16-element f16 fragment as two 16-byte chunks.
__device__ __forceinline__ v16h ld_frag(const _Float16* p0, const _Float16* p1) {
    union { f32x4 f[2]; v16h v; } u;
    u.f[0] = *(const f32x4*)p0;
    u.f[1] = *(const f32x4*)p1;
    return u.v;
}

__device__ __forceinline__ v8f wmma_f32_f16(v16h a, v16h b, v8f c) {
    // D = A(16x32 f16) * B(32x16 f16) + C(16x16 f32)
    return __builtin_amdgcn_wmma_f32_16x16x32_f16(
        /*neg_a=*/false, a, /*neg_b=*/false, b,
        /*c_mod=*/(short)0, c, /*reuse_a=*/false, /*reuse_b=*/false);
}

#if USE_TDM
// Issue a TDM 2D tile load: tile_rows rows of tile_k f16 elements (contiguous),
// row stride = stride_elems, into LDS at lds_addr.  D# layout per CDNA5 ISA §8.
__device__ __forceinline__ void tdm_load_tile_f16(unsigned int lds_addr,
                                                  const void* gptr,
                                                  unsigned int tile_k,
                                                  unsigned int tile_rows,
                                                  unsigned long long stride_elems)
{
    const unsigned long long ga = (unsigned long long)(uintptr_t)gptr;
    u32x4 g0 = {0u, 0u, 0u, 0u};
    g0.s0 = 1u;                                    // count=1, user descriptor
    g0.s1 = lds_addr;                              // LDS byte address
    g0.s2 = (unsigned int)ga;                      // global_addr[31:0]
    g0.s3 = (unsigned int)(ga >> 32) | (2u << 30); // global_addr[56:32] | type=2

    const unsigned int td0 = 0x40000000u;          // tensor_dim0 (no OOB clip)
    const unsigned int td1 = 0x40000000u;          // tensor_dim1
    i32x8 g1 = {0, 0, 0, 0, 0, 0, 0, 0};
    g1.s0 = (int)(1u << 16);                       // data_size=1 -> 2 bytes/elem
    g1.s1 = (int)((td0 & 0xFFFFu) << 16);                       // dim0[15:0]
    g1.s2 = (int)((td0 >> 16) | ((td1 & 0xFFFFu) << 16));       // dim0hi|dim1lo
    g1.s3 = (int)((td1 >> 16) | ((tile_k & 0xFFFFu) << 16));    // dim1hi|tile0
    g1.s4 = (int)(tile_rows & 0xFFFFu);                         // tile1 (tile2=0)
    g1.s5 = (int)(unsigned int)stride_elems;                    // stride0[31:0]
    g1.s6 = (int)(unsigned int)((stride_elems >> 32) & 0xFFFFu);// stride0[47:32]
    g1.s7 = 0;

    i32x4 gz = {0, 0, 0, 0};
#if defined(__clang_major__) && __clang_major__ >= 23
    i32x8 gz8 = {0, 0, 0, 0, 0, 0, 0, 0};
    __builtin_amdgcn_tensor_load_to_lds(g0, g1, gz, gz, gz8, 0);
#else
    __builtin_amdgcn_tensor_load_to_lds(g0, g1, gz, gz, 0);
#endif
}
#endif // USE_TDM

enum StoreMode {
    ST_F32       = 0,  // Cf[M,N] fp32
    ST_F16       = 1,  // Ch[M,N] f16
    ST_F16_RELU  = 2,  // Ch[M,N] f16, relu
    ST_F16_HEADS = 3,  // Ch[B,H,S,HD] f16 (split heads)
    ST_F16_VT    = 4   // Ch[B,H,HD,S] f16 (split heads + transpose)
};

// ---------------------------------------------------------------------------
// GEMM: C[M,N] = A[M,K] * Bt[N,K]^T (+bias).  Block tile 128x128; 8 waves in
// a 4(M)x2(N) arrangement; each wave computes 32x64 with 8 WMMA accumulators
// (8 WMMAs per K-step between barriers).  K-step 32, double-buffered LDS
// staging filled by the Tensor Data Mover (async DMA prefetch).
// ---------------------------------------------------------------------------
__global__ __launch_bounds__(256)
void gemm_wmma_f16(const _Float16* __restrict__ A,
                   const _Float16* __restrict__ Bt,
                   const float*    __restrict__ bias,
                   float*          __restrict__ Cf,
                   _Float16*       __restrict__ Ch,
                   int M, int N, int K,
                   int mode, int S, int H, int HD)
{
    __shared__ __align__(16) _Float16 ldsA[2][128 * 32];  // 8KB x2
    __shared__ __align__(16) _Float16 ldsB[2][128 * 32];  // 8KB x2

    const int tid  = threadIdx.x;
    const int wid  = tid >> 5;
    const int lane = tid & 31;
    const int mlo  = lane & 15;
    const int hi   = lane >> 4;
    const int wm   = wid & 3;           // wave row   (4 in M)
    const int wn   = wid >> 2;          // wave column(2 in N)

    const int mrow0 = wm * 32;          // wave's rows within block tile
    const int ncol0 = wn * 64;          // wave's cols within block tile
    const int kIter = K >> 5;

    const _Float16* Ablk = A  + (size_t)(blockIdx.x * 128) * K;
    const _Float16* Bblk = Bt + (size_t)(blockIdx.y * 128) * K;

    v8f acc[2][4] = {};

#if USE_TDM
    // Prologue: DMA tile 0 into buffer 0.
    if (wid == 0) {
        tdm_load_tile_f16((unsigned int)(uintptr_t)&ldsA[0][0], Ablk, 32, 128, (unsigned long long)K);
        tdm_load_tile_f16((unsigned int)(uintptr_t)&ldsB[0][0], Bblk, 32, 128, (unsigned long long)K);
    }
    for (int it = 0; it < kIter; ++it) {
        const int cur = it & 1;
        if (wid == 0) __builtin_amdgcn_s_wait_tensorcnt(0);
        __syncthreads();                       // tile[cur] ready for all waves
        if (wid == 0 && (it + 1) < kIter) {    // prefetch next K-slice (async)
            const int nk = (it + 1) << 5;
            tdm_load_tile_f16((unsigned int)(uintptr_t)&ldsA[cur ^ 1][0], Ablk + nk, 32, 128, (unsigned long long)K);
            tdm_load_tile_f16((unsigned int)(uintptr_t)&ldsB[cur ^ 1][0], Bblk + nk, 32, 128, (unsigned long long)K);
        }
        v16h a[2];
#pragma unroll
        for (int mi = 0; mi < 2; ++mi) {
            const _Float16* aw = &ldsA[cur][(mrow0 + mi * 16 + mlo) * 32];
            a[mi] = ld_frag(aw + hi * 8, aw + 16 + hi * 8);
        }
#pragma unroll
        for (int t = 0; t < 4; ++t) {
            const _Float16* bw = &ldsB[cur][(ncol0 + t * 16 + mlo) * 32 + hi * 8];
            v16h b = ld_frag(bw, bw + 16);
            acc[0][t] = wmma_f32_f16(a[0], b, acc[0][t]);
            acc[1][t] = wmma_f32_f16(a[1], b, acc[1][t]);
        }
        __syncthreads();                       // done reading tile[cur]
    }
#else
    // Fallback: cooperative staging with plain loads (also keeps host pass happy).
    for (int it = 0; it < kIter; ++it) {
        const int cur = it & 1;
        const int k0  = it << 5;
        {
            const int r = tid >> 1, c = (tid & 1) * 16;   // A: 128 rows x 32
            *(f32x4*)&ldsA[cur][r * 32 + c]     = *(const f32x4*)(Ablk + (size_t)r * K + k0 + c);
            *(f32x4*)&ldsA[cur][r * 32 + c + 8] = *(const f32x4*)(Ablk + (size_t)r * K + k0 + c + 8);
            *(f32x4*)&ldsB[cur][r * 32 + c]     = *(const f32x4*)(Bblk + (size_t)r * K + k0 + c);
            *(f32x4*)&ldsB[cur][r * 32 + c + 8] = *(const f32x4*)(Bblk + (size_t)r * K + k0 + c + 8);
        }
        __syncthreads();
        v16h a[2];
#pragma unroll
        for (int mi = 0; mi < 2; ++mi) {
            const _Float16* aw = &ldsA[cur][(mrow0 + mi * 16 + mlo) * 32];
            a[mi] = ld_frag(aw + hi * 8, aw + 16 + hi * 8);
        }
#pragma unroll
        for (int t = 0; t < 4; ++t) {
            const _Float16* bw = &ldsB[cur][(ncol0 + t * 16 + mlo) * 32 + hi * 8];
            v16h b = ld_frag(bw, bw + 16);
            acc[0][t] = wmma_f32_f16(a[0], b, acc[0][t]);
            acc[1][t] = wmma_f32_f16(a[1], b, acc[1][t]);
        }
        __syncthreads();
    }
#endif

    // Epilogue: lane l, vgpr r: row = base + r + hi*8, col = base + t*16 + mlo
#pragma unroll
    for (int mi = 0; mi < 2; ++mi) {
#pragma unroll
        for (int t = 0; t < 4; ++t) {
            const int col = blockIdx.y * 128 + ncol0 + t * 16 + mlo;
            const float bv = bias ? bias[col] : 0.0f;
#pragma unroll
            for (int r = 0; r < 8; ++r) {
                const int row = blockIdx.x * 128 + mrow0 + mi * 16 + r + hi * 8;
                float v = acc[mi][t][r] + bv;
                if (mode == ST_F32) {
                    Cf[(size_t)row * N + col] = v;
                } else if (mode == ST_F16) {
                    Ch[(size_t)row * N + col] = (_Float16)v;
                } else if (mode == ST_F16_RELU) {
                    Ch[(size_t)row * N + col] = (_Float16)fmaxf(v, 0.0f);
                } else if (mode == ST_F16_HEADS) {
                    const int b = row / S, s = row % S;
                    const int h = col / HD, d = col % HD;
                    Ch[(((size_t)b * H + h) * S + s) * HD + d] = (_Float16)v;
                } else { // ST_F16_VT
                    const int b = row / S, s = row % S;
                    const int h = col / HD, d = col % HD;
                    Ch[(((size_t)b * H + h) * HD + d) * S + s] = (_Float16)v;
                }
            }
        }
    }
}

// ---------------------------------------------------------------------------
// Flash attention: grid (S/128, B*H); 8 waves/block; wave owns 16 query rows.
// Q,K: [B,H,S,HD] f16.  Vt: [B,H,HD,S] f16.  O: [B,S,D] f16 (merged heads).
// ---------------------------------------------------------------------------
__global__ __launch_bounds__(256)
void flash_attn_wmma(const _Float16* __restrict__ Q,
                     const _Float16* __restrict__ Kk,
                     const _Float16* __restrict__ Vt,
                     _Float16*       __restrict__ O,
                     int S, int H, int HD, float scale)
{
    __shared__ __align__(16) _Float16 plds[8 * 16 * 32]; // per-wave P tile

    const int tid  = threadIdx.x;
    const int wid  = tid >> 5;
    const int lane = tid & 31;
    const int mlo  = lane & 15;
    const int hi   = lane >> 4;

    const int bh = blockIdx.y;
    const int q0 = blockIdx.x * 128 + wid * 16;

    const _Float16* Qb = Q  + (size_t)bh * S * HD;
    const _Float16* Kb = Kk + (size_t)bh * S * HD;
    const _Float16* Vb = Vt + (size_t)bh * HD * S;
    _Float16* pw = plds + wid * (16 * 32);

    // Q tile (16 x 64) as two A fragments; stays in registers for all of S.
    v16h qa[2];
#pragma unroll
    for (int kc = 0; kc < 2; ++kc) {
        const _Float16* qr = Qb + (size_t)(q0 + mlo) * HD + kc * 32;
        qa[kc] = ld_frag(qr + hi * 8, qr + 16 + hi * 8);
    }

    v8f o[4] = {};
    float mrow[8], lrow[8];
#pragma unroll
    for (int r = 0; r < 8; ++r) { mrow[r] = -3.0e38f; lrow[r] = 0.0f; }

    for (int j = 0; j < S; j += 32) {
        // ---- scores: two 16x16 tiles over the 32-key chunk ----
        v8f st[2];
#pragma unroll
        for (int nt = 0; nt < 2; ++nt) {
            v8f s = {};
#pragma unroll
            for (int kc = 0; kc < 2; ++kc) {
                const _Float16* kr = Kb + (size_t)(j + nt * 16 + mlo) * HD + kc * 32 + hi * 8;
                v16h kb = ld_frag(kr, kr + 16);
                s = wmma_f32_f16(qa[kc], kb, s);
            }
#pragma unroll
            for (int r = 0; r < 8; ++r) st[nt][r] = s[r] * scale;
        }

        // ---- prefetch V fragments now; latency hides under softmax VALU ----
        v16h vbf[4];
#pragma unroll
        for (int t = 0; t < 4; ++t) {
            const _Float16* vr = Vb + (size_t)(t * 16 + mlo) * S + j + hi * 8;
            vbf[t] = ld_frag(vr, vr + 16);
        }

        // ---- online softmax (row stats replicated across 16-lane groups) ----
#pragma unroll
        for (int r = 0; r < 8; ++r) {
            float tmax = fmaxf(st[0][r], st[1][r]);
#pragma unroll
            for (int off = 8; off; off >>= 1)
                tmax = fmaxf(tmax, __shfl_xor(tmax, off));
            const float mnew = fmaxf(mrow[r], tmax);
            const float al   = __expf(mrow[r] - mnew);
            const float p0   = __expf(st[0][r] - mnew);
            const float p1   = __expf(st[1][r] - mnew);
            st[0][r] = p0; st[1][r] = p1;
            float ps = p0 + p1;
#pragma unroll
            for (int off = 8; off; off >>= 1)
                ps += __shfl_xor(ps, off);
            lrow[r] = lrow[r] * al + ps;
            mrow[r] = mnew;
#pragma unroll
            for (int t = 0; t < 4; ++t) o[t][r] *= al;
        }

        // ---- convert P (C-layout f32) -> A-layout f16 via per-wave LDS ----
#pragma unroll
        for (int nt = 0; nt < 2; ++nt)
#pragma unroll
            for (int r = 0; r < 8; ++r)
                pw[(r + hi * 8) * 32 + nt * 16 + mlo] = (_Float16)st[nt][r];
        asm volatile("s_wait_dscnt 0" ::: "memory");
        v16h pa = ld_frag(pw + mlo * 32 + hi * 8, pw + mlo * 32 + 16 + hi * 8);

        // ---- O += P (16x32) * V (32x64) ----
#pragma unroll
        for (int t = 0; t < 4; ++t)
            o[t] = wmma_f32_f16(pa, vbf[t], o[t]);
    }

    // ---- normalize + store merged heads [B,S,D] f16 ----
    const int b = bh / H, h = bh % H;
    const int D = H * HD;
#pragma unroll
    for (int r = 0; r < 8; ++r) {
        const float inv = 1.0f / lrow[r];
        const int row = q0 + r + hi * 8;
#pragma unroll
        for (int t = 0; t < 4; ++t) {
            O[((size_t)(b * S + row)) * D + h * HD + t * 16 + mlo] =
                (_Float16)(o[t][r] * inv);
        }
    }
}

// ---------------------------------------------------------------------------
// Fused residual-add + LayerNorm over D=1024; writes f32 and (optional) f16.
// ---------------------------------------------------------------------------
__global__ __launch_bounds__(256)
void add_layernorm(const float* __restrict__ X, const float* __restrict__ Y,
                   const float* __restrict__ g, const float* __restrict__ bta,
                   float* __restrict__ outF, _Float16* __restrict__ outH, int D)
{
    const int row = blockIdx.x;
    const int tid = threadIdx.x;
    const float* xr = X + (size_t)row * D;
    const float* yr = Y + (size_t)row * D;

    __shared__ float redS[8], redQ[8];

    float vals[4];
    float s = 0.0f, ss = 0.0f;
#pragma unroll
    for (int i = 0; i < 4; ++i) {               // D = 1024 = 4*256
        const float v = xr[tid + i * 256] + yr[tid + i * 256];
        vals[i] = v; s += v; ss += v * v;
    }
#pragma unroll
    for (int off = 16; off; off >>= 1) {
        s  += __shfl_xor(s, off);
        ss += __shfl_xor(ss, off);
    }
    const int wid = tid >> 5, lane = tid & 31;
    if (lane == 0) { redS[wid] = s; redQ[wid] = ss; }
    __syncthreads();
    s = 0.0f; ss = 0.0f;
#pragma unroll
    for (int w = 0; w < 8; ++w) { s += redS[w]; ss += redQ[w]; }

    const float mu  = s / (float)D;
    const float var = ss / (float)D - mu * mu;
    const float inv = rsqrtf(var + 1e-5f);
#pragma unroll
    for (int i = 0; i < 4; ++i) {
        const int c = tid + i * 256;
        const float v = (vals[i] - mu) * inv * g[c] + bta[c];
        outF[(size_t)row * D + c] = v;
        if (outH) outH[(size_t)row * D + c] = (_Float16)v;
    }
}

// ---------------------------------------------------------------------------
// Prep kernels: f32 -> f16 convert, and f32 [K,N] -> f16 [N,K] transpose.
// ---------------------------------------------------------------------------
__global__ void f32_to_f16_kernel(const float* __restrict__ in,
                                  _Float16* __restrict__ out, size_t n)
{
    for (size_t i = (size_t)blockIdx.x * blockDim.x + threadIdx.x; i < n;
         i += (size_t)gridDim.x * blockDim.x)
        out[i] = (_Float16)in[i];
}

__global__ void transpose_f32_to_f16(const float* __restrict__ W,
                                     _Float16* __restrict__ Wt, int K, int N)
{
    const size_t total = (size_t)K * N;
    for (size_t i = (size_t)blockIdx.x * blockDim.x + threadIdx.x; i < total;
         i += (size_t)gridDim.x * blockDim.x) {
        const size_t n = i / K, k = i % K;     // Wt is [N][K]
        Wt[i] = (_Float16)W[k * (size_t)N + n];
    }
}

// ---------------------------------------------------------------------------
// Orchestration
// ---------------------------------------------------------------------------
extern "C" void kernel_launch(void* const* d_in, const int* in_sizes, int n_in,
                              void* d_out, int out_size, void* d_ws, size_t ws_size,
                              hipStream_t stream)
{
    (void)in_sizes; (void)n_in; (void)out_size; (void)ws_size;

    constexpr int B = 2, S = 2048, D = 1024, H = 16, HD = 64, DFF = 4096;
    constexpr int M = B * S;                       // 4096 rows
    const float scale = 1.05f / 8.0f;              // GEOM / sqrt(HD)

    const float* x     = (const float*)d_in[0];
    const float* Wq    = (const float*)d_in[1];
    const float* bq    = (const float*)d_in[2];
    const float* Wk    = (const float*)d_in[3];
    const float* bk    = (const float*)d_in[4];
    const float* Wv    = (const float*)d_in[5];
    const float* bv    = (const float*)d_in[6];
    const float* Wo    = (const float*)d_in[7];
    const float* bo    = (const float*)d_in[8];
    const float* g1    = (const float*)d_in[9];
    const float* b1    = (const float*)d_in[10];
    const float* Wff1  = (const float*)d_in[11];
    const float* bff1  = (const float*)d_in[12];
    const float* Wff2  = (const float*)d_in[13];
    const float* bff2  = (const float*)d_in[14];
    const float* g2    = (const float*)d_in[15];
    const float* b2    = (const float*)d_in[16];
    float* out = (float*)d_out;

    // ---- workspace carve (256B aligned) ----
    char* w = (char*)d_ws;
    auto carve = [&](size_t bytes) -> void* {
        void* p = (void*)w;
        w += (bytes + 255) & ~(size_t)255;
        return p;
    };
    _Float16* Xh     = (_Float16*)carve((size_t)M * D * 2);
    _Float16* WqT    = (_Float16*)carve((size_t)D * D * 2);
    _Float16* WkT    = (_Float16*)carve((size_t)D * D * 2);
    _Float16* WvT    = (_Float16*)carve((size_t)D * D * 2);
    _Float16* WoT    = (_Float16*)carve((size_t)D * D * 2);
    _Float16* Wff1T  = (_Float16*)carve((size_t)D * DFF * 2);
    _Float16* Wff2T  = (_Float16*)carve((size_t)DFF * D * 2);
    _Float16* Qh     = (_Float16*)carve((size_t)M * D * 2);
    _Float16* Kh     = (_Float16*)carve((size_t)M * D * 2);
    _Float16* Vth    = (_Float16*)carve((size_t)M * D * 2);
    _Float16* attnH  = (_Float16*)carve((size_t)M * D * 2);
    _Float16* ffh    = (_Float16*)carve((size_t)M * DFF * 2);
    _Float16* h1h    = (_Float16*)carve((size_t)M * D * 2);
    float*    attnF  = (float*)carve((size_t)M * D * 4);
    float*    h1f    = (float*)carve((size_t)M * D * 4);
    float*    ffF    = (float*)carve((size_t)M * D * 4);

    // ---- prep: f16 copies / weight transposes (tiny vs GEMM cost) ----
    f32_to_f16_kernel<<<2048, 256, 0, stream>>>(x, Xh, (size_t)M * D);
    transpose_f32_to_f16<<<2048, 256, 0, stream>>>(Wq,   WqT,   D,   D);
    transpose_f32_to_f16<<<2048, 256, 0, stream>>>(Wk,   WkT,   D,   D);
    transpose_f32_to_f16<<<2048, 256, 0, stream>>>(Wv,   WvT,   D,   D);
    transpose_f32_to_f16<<<2048, 256, 0, stream>>>(Wo,   WoT,   D,   D);
    transpose_f32_to_f16<<<4096, 256, 0, stream>>>(Wff1, Wff1T, D,   DFF);
    transpose_f32_to_f16<<<4096, 256, 0, stream>>>(Wff2, Wff2T, DFF, D);

    const dim3 blk(256);
    // ---- QKV projections (WMMA + TDM prefetch) ----
    {
        dim3 grid(M / 128, D / 128);
        gemm_wmma_f16<<<grid, blk, 0, stream>>>(Xh, WqT, bq, nullptr, Qh,
                                                M, D, D, ST_F16_HEADS, S, H, HD);
        gemm_wmma_f16<<<grid, blk, 0, stream>>>(Xh, WkT, bk, nullptr, Kh,
                                                M, D, D, ST_F16_HEADS, S, H, HD);
        gemm_wmma_f16<<<grid, blk, 0, stream>>>(Xh, WvT, bv, nullptr, Vth,
                                                M, D, D, ST_F16_VT, S, H, HD);
    }
    // ---- flash attention (WMMA QK^T and P*V) ----
    {
        dim3 grid(S / 128, B * H);
        flash_attn_wmma<<<grid, blk, 0, stream>>>(Qh, Kh, Vth, attnH, S, H, HD, scale);
    }
    // ---- output projection ----
    {
        dim3 grid(M / 128, D / 128);
        gemm_wmma_f16<<<grid, blk, 0, stream>>>(attnH, WoT, bo, attnF, nullptr,
                                                M, D, D, ST_F32, S, H, HD);
    }
    // ---- LN1 (residual + attn) ----
    add_layernorm<<<M, blk, 0, stream>>>(x, attnF, g1, b1, h1f, h1h, D);
    // ---- FFN ----
    {
        dim3 grid(M / 128, DFF / 128);
        gemm_wmma_f16<<<grid, blk, 0, stream>>>(h1h, Wff1T, bff1, nullptr, ffh,
                                                M, DFF, D, ST_F16_RELU, S, H, HD);
    }
    {
        dim3 grid(M / 128, D / 128);
        gemm_wmma_f16<<<grid, blk, 0, stream>>>(ffh, Wff2T, bff2, ffF, nullptr,
                                                M, D, DFF, ST_F32, S, H, HD);
    }
    // ---- LN2 -> output (f32) ----
    add_layernorm<<<M, blk, 0, stream>>>(h1f, ffF, g2, b2, out, nullptr, D);
}
